// Attention_65609920413963
// MI455X (gfx1250) — matrix operand order repaired and checked by reading, not compiled, not run on hardware
//
#include <hip/hip_runtime.h>
#include <hip/hip_bf16.h>

#define B_  4
#define N_  1024
#define C_  1024
#define H_  16
#define DH  64
#define NEGV (-1e10f)
#define L2E 1.44269504088896340736f

typedef __attribute__((ext_vector_type(16))) __bf16 v16bf;
typedef __attribute__((ext_vector_type(8)))  __bf16 bf16x8;
typedef __attribute__((ext_vector_type(8)))  float  v8f;
typedef __attribute__((ext_vector_type(8)))  float  f32x8;

static __device__ inline v8f wmma_bf16(v16bf a, v16bf b, v8f c) {
  return __builtin_amdgcn_wmma_f32_16x16x32_bf16(false, a, false, b, (short)0, c,
                                                 false, false);
}

// A-frag (16x32, bf16 K-contiguous source, row stride ld elements).
static __device__ inline v16bf load_a_frag_bf16(const __bf16* base, int ld,
                                                int m0, int k0, int lane) {
  const int hl = lane >> 4;
  const __bf16* p = base + (size_t)(m0 + (lane & 15)) * ld + k0 + 16 * hl;
  bf16x8 lo = *(const bf16x8*)p;
  bf16x8 hi = *(const bf16x8*)(p + 8);
  return __builtin_shufflevector(lo, hi, 0, 1, 2, 3, 4, 5, 6, 7,
                                 8, 9, 10, 11, 12, 13, 14, 15);
}

static __device__ inline v16bf load_a_frag_f32(const float* base, int ld,
                                               int m0, int k0, int lane) {
  const int hl = lane >> 4;
  const float* p = base + (size_t)(m0 + (lane & 15)) * ld + k0 + 16 * hl;
  f32x8 lo = *(const f32x8*)p;
  f32x8 hi = *(const f32x8*)(p + 8);
  v16bf a;
#pragma unroll
  for (int i = 0; i < 8; ++i) { a[i] = (__bf16)lo[i]; a[i + 8] = (__bf16)hi[i]; }
  return a;
}

// B-frag (32x16): source S[n, k] row-major (k contiguous), two 16B reads.
static __device__ inline v16bf load_b_frag_bf16(const __bf16* base, int ld,
                                                int n0, int k0, int lane) {
  const __bf16* p =
      base + (size_t)(n0 + (lane & 15)) * ld + k0 + 16 * (lane >> 4);
  bf16x8 lo = *(const bf16x8*)p;
  bf16x8 hi = *(const bf16x8*)(p + 8);
  return __builtin_shufflevector(lo, hi, 0, 1, 2, 3, 4, 5, 6, 7,
                                 8, 9, 10, 11, 12, 13, 14, 15);
}

static __device__ inline v16bf load_b_frag_f32(const float* base, int ld,
                                               int n0, int k0, int lane) {
  const float* p =
      base + (size_t)(n0 + (lane & 15)) * ld + k0 + 16 * (lane >> 4);
  v16bf b;
#pragma unroll
  for (int i = 0; i < 16; ++i) b[i] = (__bf16)p[i];
  return b;
}

// Cooperative async copy of a 64x64 bf16 tile (global row stride gld elems)
// into LDS with padded row stride 72 elems. 128 threads x 4 b128 chunks.
static __device__ __attribute__((always_inline)) void stage_tile_async(
    const __bf16* gbase, int gld, __bf16* lbase, int tid) {
#pragma unroll
  for (int i = 0; i < 4; ++i) {
    const int c = tid + 128 * i;           // 0..511 chunk id
    const int row = c >> 3;                // 0..63
    const int col = (c & 7) * 8;           // 0..56 (8 bf16 = 16B)
    const __bf16* g = gbase + (size_t)row * gld + col;
    const unsigned loff = (unsigned)(size_t)(lbase + row * 72 + col);
    asm volatile("global_load_async_to_lds_b128 %0, %1, off"
                 :: "v"(loff), "v"(g) : "memory");
  }
}

#if __has_builtin(__builtin_amdgcn_tensor_load_to_lds)
#define HAVE_TDM 1
typedef unsigned int u32x4 __attribute__((ext_vector_type(4)));
typedef int i32x8 __attribute__((ext_vector_type(8)));
typedef int i32x4 __attribute__((ext_vector_type(4)));

// TDM: load a 64x64 bf16 tile (row stride dim0_stride elems) into LDS with
// 16B padding after every 128B row (LDS row stride 144B = 72 elems).
static __device__ __attribute__((always_inline)) void tdm_load_tile(
    const __bf16* g, unsigned lds_off, unsigned dim0_stride) {
  const unsigned long long ga = (unsigned long long)(size_t)g;
  u32x4 g0;
  g0[0] = 1u;                                   // count=1 valid user D#
  g0[1] = lds_off;                              // lds_addr (bytes)
  g0[2] = (unsigned)ga;                         // global_addr[31:0]
  g0[3] = ((unsigned)(ga >> 32) & 0x01ffffffu)  // global_addr[56:32]
          | 0x80000000u;                        // type=2 ("image")
  i32x8 g1;
  g1[0] = (int)((1u << 16)      // data_size = 2 bytes
                | (1u << 20)    // pad_enable
                | (4u << 22)    // pad_interval: 32 DWORDs (128B)
                | (3u << 25));  // pad_amount: 4 DWORDs (16B)
  g1[1] = (int)(64u << 16);     // tensor_dim0 = 64
  g1[2] = (int)(64u << 16);     // tensor_dim1 = 64 (>= tile rows)
  g1[3] = (int)(64u << 16);     // tile_dim0 = 64
  g1[4] = 64;                   // tile_dim1 = 64
  g1[5] = (int)dim0_stride;     // tensor_dim0_stride (elems)
  g1[6] = 0;
  g1[7] = 0;
  i32x4 z4 = {0, 0, 0, 0};
  i32x8 z8 = {0, 0, 0, 0, 0, 0, 0, 0};
  // 6-arg form (clang-23 / therock-10.0 headers), zero groups 2/3 for 2D tile
  __builtin_amdgcn_tensor_load_to_lds(g0, g1, z4, z4, z8, 0);
}
#endif

// ---------------------------------------------------------------------------
// Kernel 1: qkv = x @ qkv_w.T + qkv_b, scattered into bf16 Q[B,H,N,64],
// K[B,H,N,64], Vt[B,H,64,N] (V transposed for P@V B-frags).
// ---------------------------------------------------------------------------
__global__ __launch_bounds__(128) void qkv_wmma_kernel(
    const float* __restrict__ x, const float* __restrict__ w,
    const float* __restrict__ bias, __bf16* __restrict__ qb,
    __bf16* __restrict__ kb, __bf16* __restrict__ vT) {
  const int lane = threadIdx.x & 31;
  const int wv = threadIdx.x >> 5;
  const int m0 = blockIdx.y * 64 + (wv >> 1) * 32;
  const int n0 = blockIdx.x * 64 + (wv & 1) * 32;
  v8f acc[2][2] = {};
  for (int k0 = 0; k0 < C_; k0 += 32) {
    if (k0 + 32 < C_) {
      __builtin_prefetch(x + (size_t)(m0 + (lane & 15)) * C_ + k0 + 32, 0, 1);
      __builtin_prefetch(w + (size_t)(n0 + (lane & 15)) * C_ + k0 + 32, 0, 1);
    }
    v16bf a0 = load_a_frag_f32(x, C_, m0, k0, lane);
    v16bf a1 = load_a_frag_f32(x, C_, m0 + 16, k0, lane);
    v16bf b0 = load_b_frag_f32(w, C_, n0, k0, lane);
    v16bf b1 = load_b_frag_f32(w, C_, n0 + 16, k0, lane);
    acc[0][0] = wmma_bf16(a0, b0, acc[0][0]);
    acc[0][1] = wmma_bf16(a0, b1, acc[0][1]);
    acc[1][0] = wmma_bf16(a1, b0, acc[1][0]);
    acc[1][1] = wmma_bf16(a1, b1, acc[1][1]);
  }
  const int hl = lane >> 4, cl = lane & 15;
#pragma unroll
  for (int j = 0; j < 2; ++j) {
    const int nn = n0 + j * 16 + cl;
    const float bj = bias[nn];
#pragma unroll
    for (int i = 0; i < 2; ++i) {
#pragma unroll
      for (int r = 0; r < 8; ++r) {
        const int m = m0 + i * 16 + r + 8 * hl;
        const int bb = m >> 10, n = m & (N_ - 1);
        const __bf16 bv = (__bf16)(acc[i][j][r] + bj);
        if (nn < C_) {
          const int h = nn >> 6, d = nn & 63;
          qb[(((size_t)bb * H_ + h) * N_ + n) * DH + d] = bv;
        } else if (nn < 2 * C_) {
          const int t = nn - C_, h = t >> 6, d = t & 63;
          kb[(((size_t)bb * H_ + h) * N_ + n) * DH + d] = bv;
        } else {
          const int t = nn - 2 * C_, h = t >> 6, d = t & 63;
          vT[(((size_t)bb * H_ + h) * DH + d) * N_ + n] = bv;
        }
      }
    }
  }
}

// ---------------------------------------------------------------------------
// Kernel 2: fused masked-bias attention (flash style, online softmax).
// K tiles staged by the Tensor Data Mover (TENSORcnt), V tiles by async
// global->LDS b128 copies (ASYNCcnt); both double-buffered.
// ---------------------------------------------------------------------------
__global__ __launch_bounds__(128) void flash_attn_kernel(
    const __bf16* __restrict__ qb, const __bf16* __restrict__ kb,
    const __bf16* __restrict__ vT, const float* __restrict__ Kmat,
    const int* __restrict__ n1, const int* __restrict__ n2,
    __bf16* __restrict__ ao) {
  __shared__ __align__(16) __bf16 ksh[2][64][72];     // K tile, TDM-filled
  __shared__ __align__(16) __bf16 vsh[2][64][72];     // V tile (d-major)
  __shared__ __align__(16) __bf16 pshare[4][16][72];  // per-wave P tile
  const int tid = threadIdx.x;
  const int lane = tid & 31;
  const int wv = tid >> 5;
  const int b = blockIdx.z, h = blockIdx.y;
  const int row0 = blockIdx.x * 64 + wv * 16;
  const int L = n1[b] * n2[b];
  const float scale = 0.125f;  // 1/sqrt(64)
  const int hl = lane >> 4, cl = lane & 15;

  const __bf16* qbh = qb + (size_t)(b * H_ + h) * N_ * DH;
  const __bf16* kbh = kb + (size_t)(b * H_ + h) * N_ * DH;
  const __bf16* vth = vT + (size_t)(b * H_ + h) * DH * N_;
  const float* Kb = Kmat + (size_t)b * N_ * N_;

  v16bf qf[2];
#pragma unroll
  for (int c = 0; c < 2; ++c) qf[c] = load_a_frag_bf16(qbh, DH, row0, c * 32, lane);

  v8f o[4] = {};
  float rmax[8], rsum[8];
#pragma unroll
  for (int r = 0; r < 8; ++r) { rmax[r] = -3.0e38f; rsum[r] = 0.f; }

  __bf16* pw = &pshare[wv][0][0];

  // prologue: stage first K/V tiles into buffer 0
#ifdef HAVE_TDM
  if (wv == 0)
    tdm_load_tile(kbh, (unsigned)(size_t)&ksh[0][0][0], DH);
#else
  stage_tile_async(kbh, DH, &ksh[0][0][0], tid);
#endif
  stage_tile_async(vth, N_, &vsh[0][0][0], tid);

  int cur = 0;
  for (int jc = 0; jc < N_; jc += 64) {
    __builtin_amdgcn_s_wait_tensorcnt(0);            // no-op for waves 1..3
    asm volatile("s_wait_asynccnt 0x0" ::: "memory");
    __syncthreads();  // tiles for jc visible; previous buffer fully consumed
    if (jc + 64 < N_) {  // prefetch next block into the other buffer
#ifdef HAVE_TDM
      if (wv == 0)
        tdm_load_tile(kbh + (size_t)(jc + 64) * DH,
                      (unsigned)(size_t)&ksh[cur ^ 1][0][0], DH);
#else
      stage_tile_async(kbh + (size_t)(jc + 64) * DH, DH, &ksh[cur ^ 1][0][0], tid);
#endif
      stage_tile_async(vth + (jc + 64), N_, &vsh[cur ^ 1][0][0], tid);
    }
    // scores S = q @ k^T over this 64-key block (K tile from LDS)
    v8f s[4] = {};
#pragma unroll
    for (int t = 0; t < 4; ++t) {
#pragma unroll
      for (int c = 0; c < 2; ++c) {
        v16bf bkf = load_b_frag_bf16(&ksh[cur][0][0], 72, t * 16, c * 32, lane);
        s[t] = wmma_bf16(qf[c], bkf, s[t]);
      }
    }
    // logits = (masked ? NEG : scale*S) + K[b]
    float sv[4][8], bmax[8];
#pragma unroll
    for (int r = 0; r < 8; ++r) bmax[r] = -3.0e38f;
#pragma unroll
    for (int t = 0; t < 4; ++t) {
      const int col = jc + t * 16 + cl;
      const bool coob = col >= L;
#pragma unroll
      for (int r = 0; r < 8; ++r) {
        const int row = row0 + r + 8 * hl;
        const float kadd = Kb[(size_t)row * N_ + col];
        float v = ((coob || row >= L) ? NEGV : (scale * s[t][r])) + kadd;
        sv[t][r] = v;
        bmax[r] = fmaxf(bmax[r], v);
      }
    }
    // online softmax row update (rows live in groups of 16 lanes per half)
#pragma unroll
    for (int r = 0; r < 8; ++r) {
      float m = bmax[r];
      m = fmaxf(m, __shfl_xor(m, 1));
      m = fmaxf(m, __shfl_xor(m, 2));
      m = fmaxf(m, __shfl_xor(m, 4));
      m = fmaxf(m, __shfl_xor(m, 8));
      const float newm = fmaxf(rmax[r], m);
      const float alpha = exp2f((rmax[r] - newm) * L2E);
      rmax[r] = newm;
      rsum[r] *= alpha;
#pragma unroll
      for (int t = 0; t < 4; ++t) o[t][r] *= alpha;
      float psum = 0.f;
#pragma unroll
      for (int t = 0; t < 4; ++t) {
        const float p = exp2f((sv[t][r] - newm) * L2E);
        sv[t][r] = p;
        psum += p;
      }
      psum += __shfl_xor(psum, 1);
      psum += __shfl_xor(psum, 2);
      psum += __shfl_xor(psum, 4);
      psum += __shfl_xor(psum, 8);
      rsum[r] += psum;
    }
    // spill P (C-layout) to per-wave LDS, re-read in A-layout (in-order DS)
#pragma unroll
    for (int t = 0; t < 4; ++t)
#pragma unroll
      for (int r = 0; r < 8; ++r)
        pw[(r + 8 * hl) * 72 + t * 16 + cl] = (__bf16)sv[t][r];
    // O += P @ V  (V tile from LDS)
#pragma unroll
    for (int c = 0; c < 2; ++c) {
      v16bf pf = load_a_frag_bf16(pw, 72, 0, c * 32, lane);
#pragma unroll
      for (int t = 0; t < 4; ++t) {
        v16bf bvf = load_b_frag_bf16(&vsh[cur][0][0], 72, t * 16, c * 32, lane);
        o[t] = wmma_bf16(pf, bvf, o[t]);
      }
    }
    cur ^= 1;
  }
  // normalize and store to attnout[B,N,C] bf16
#pragma unroll
  for (int t = 0; t < 4; ++t) {
#pragma unroll
    for (int r = 0; r < 8; ++r) {
      const int row = row0 + r + 8 * hl;
      const int d = t * 16 + cl;
      ao[((size_t)b * N_ + row) * C_ + h * DH + d] = (__bf16)(o[t][r] / rsum[r]);
    }
  }
}

// ---------------------------------------------------------------------------
// Kernel 3: out = attnout @ proj_w.T + proj_b  (fp32 output)
// ---------------------------------------------------------------------------
__global__ __launch_bounds__(128) void proj_wmma_kernel(
    const __bf16* __restrict__ a, const float* __restrict__ w,
    const float* __restrict__ bias, float* __restrict__ out) {
  const int lane = threadIdx.x & 31;
  const int wv = threadIdx.x >> 5;
  const int m0 = blockIdx.y * 64 + (wv >> 1) * 32;
  const int n0 = blockIdx.x * 64 + (wv & 1) * 32;
  v8f acc[2][2] = {};
  for (int k0 = 0; k0 < C_; k0 += 32) {
    if (k0 + 32 < C_) {
      __builtin_prefetch(a + (size_t)(m0 + (lane & 15)) * C_ + k0 + 32, 0, 1);
      __builtin_prefetch(w + (size_t)(n0 + (lane & 15)) * C_ + k0 + 32, 0, 1);
    }
    v16bf a0 = load_a_frag_bf16(a, C_, m0, k0, lane);
    v16bf a1 = load_a_frag_bf16(a, C_, m0 + 16, k0, lane);
    v16bf b0 = load_b_frag_f32(w, C_, n0, k0, lane);
    v16bf b1 = load_b_frag_f32(w, C_, n0 + 16, k0, lane);
    acc[0][0] = wmma_bf16(a0, b0, acc[0][0]);
    acc[0][1] = wmma_bf16(a0, b1, acc[0][1]);
    acc[1][0] = wmma_bf16(a1, b0, acc[1][0]);
    acc[1][1] = wmma_bf16(a1, b1, acc[1][1]);
  }
  const int hl = lane >> 4, cl = lane & 15;
#pragma unroll
  for (int j = 0; j < 2; ++j) {
    const int nn = n0 + j * 16 + cl;
    const float bj = bias[nn];
#pragma unroll
    for (int i = 0; i < 2; ++i)
#pragma unroll
      for (int r = 0; r < 8; ++r) {
        const int m = m0 + i * 16 + r + 8 * hl;
        out[(size_t)m * C_ + nn] = acc[i][j][r] + bj;
      }
  }
}

extern "C" void kernel_launch(void* const* d_in, const int* in_sizes, int n_in,
                              void* d_out, int out_size, void* d_ws,
                              size_t ws_size, hipStream_t stream) {
  const float* x = (const float*)d_in[0];
  const float* Kmat = (const float*)d_in[1];
  const int* n1 = (const int*)d_in[2];
  const int* n2 = (const int*)d_in[3];
  const float* qkv_w = (const float*)d_in[4];
  const float* qkv_b = (const float*)d_in[5];
  const float* proj_w = (const float*)d_in[6];
  const float* proj_b = (const float*)d_in[7];

  const size_t QSZ = (size_t)B_ * H_ * N_ * DH;  // 4M elems
  __bf16* qb = (__bf16*)d_ws;
  __bf16* kb = qb + QSZ;
  __bf16* vT = kb + QSZ;
  __bf16* ao = vT + QSZ;  // attnout [B,N,C] bf16

  dim3 g1(3 * C_ / 64, (B_ * N_) / 64);  // (48, 64)
  qkv_wmma_kernel<<<g1, 128, 0, stream>>>(x, qkv_w, qkv_b, qb, kb, vT);

  dim3 g2(N_ / 64, H_, B_);  // (16, 16, 4)
  flash_attn_kernel<<<g2, 128, 0, stream>>>(qb, kb, vT, Kmat, n1, n2, ao);

  dim3 g3(C_ / 64, (B_ * N_) / 64);  // (16, 64)
  proj_wmma_kernel<<<g3, 128, 0, stream>>>(ao, proj_w, proj_b, (float*)d_out);
}
